// DOGMSTGATUncertaintyNet_10668698763497
// MI455X (gfx1250) — compile-verified, matrix operand order checked by reading, not run on hardware
//
#include <hip/hip_runtime.h>
#include <hip/hip_bf16.h>
#include <math.h>

#define NN 90000
#define EE 300000

typedef __attribute__((ext_vector_type(16))) _Float16 v16h;
typedef __attribute__((ext_vector_type(8)))  float    v8f;

// ---------- helpers ----------
__device__ __forceinline__ unsigned f2key(float f) {
  unsigned u = __float_as_uint(f);
  return (u & 0x80000000u) ? ~u : (u | 0x80000000u);
}
__device__ __forceinline__ float key2f(unsigned k) {
  unsigned u = (k & 0x80000000u) ? (k & 0x7FFFFFFFu) : ~k;
  return __uint_as_float(u);
}

// ---------- generic utility kernels ----------
__global__ void zero_f32_kernel(float* p, int count, float val) {
  int i = blockIdx.x * blockDim.x + threadIdx.x;
  if (i < count) p[i] = val;
}
__global__ void zero_u32_kernel(unsigned* p, int count) {
  int i = blockIdx.x * blockDim.x + threadIdx.x;
  if (i < count) p[i] = 0u;
}
__global__ void f32_to_f16_kernel(const float* __restrict__ src, _Float16* __restrict__ dst, int count) {
  int i = blockIdx.x * blockDim.x + threadIdx.x;
  if (i < count) dst[i] = (_Float16)src[i];
}

// ---------- pack weight matrix B[K,Ncol](f32) into WMMA-fragment-major f16 ----------
// Packed layout: frag = kt*ntiles + tn (kt = K/32 tiles, tn = N/16 tiles), 512 halves
// per fragment, lane-major: out[frag*512 + lane*16 + (2j+p)] = B[k(j,p,lane), n(lane)].
__global__ void pack_b_kernel(const float* __restrict__ B, _Float16* __restrict__ out,
                              int K, int Ncol) {
  int i = blockIdx.x * blockDim.x + threadIdx.x;
  if (i >= K * Ncol) return;
  int frag = i >> 9;
  int lane = (i >> 4) & 31;
  int t = i & 15;
  int j = t >> 1;
  int pp = t & 1;
  int ntiles = Ncol >> 4;
  int kt = frag / ntiles;
  int tn = frag % ntiles;
  int k = kt * 32 + ((j < 4) ? 2 * j : 2 * j + 8) + ((lane & 16) ? 8 : 0) + pp;
  int n = tn * 16 + (lane & 15);
  out[i] = (_Float16)B[k * Ncol + n];
}

// ---------- encoder: h = LN(relu(x@W+b))*g+beta, per-sensor params ----------
__global__ void encoder_kernel(const float* __restrict__ x, const int* __restrict__ sensor,
                               const float* W0, const float* b0, const float* g0, const float* be0,
                               const float* W1, const float* b1, const float* g1, const float* be1,
                               const float* W2, const float* b2, const float* g2, const float* be2,
                               _Float16* __restrict__ hout) {
  __shared__ float red[64];
  __shared__ float xr[8];
  int n = blockIdx.x;
  int j = threadIdx.x;
  if (j < 8) xr[j] = x[n * 8 + j];
  __syncthreads();
  int s = sensor[n];
  const float* W  = (s == 0) ? W0  : ((s == 1) ? W1  : W2);
  const float* b  = (s == 0) ? b0  : ((s == 1) ? b1  : b2);
  const float* g  = (s == 0) ? g0  : ((s == 1) ? g1  : g2);
  const float* be = (s == 0) ? be0 : ((s == 1) ? be1 : be2);
  float acc = b[j];
#pragma unroll
  for (int i = 0; i < 8; ++i) acc += xr[i] * W[i * 64 + j];
  acc = fmaxf(acc, 0.0f);
  red[j] = acc;
  __syncthreads();
  for (int st = 32; st > 0; st >>= 1) { if (j < st) red[j] += red[j + st]; __syncthreads(); }
  float mean = red[0] / 64.0f;
  __syncthreads();
  float d = acc - mean;
  red[j] = d * d;
  __syncthreads();
  for (int st = 32; st > 0; st >>= 1) { if (j < st) red[j] += red[j + st]; __syncthreads(); }
  float var = red[0] / 64.0f;
  float y = d * rsqrtf(var + 1e-5f) * g[j] + be[j];
  hout[n * 64 + j] = (_Float16)y;
}

// ---------- WMMA GEMM: C[M,Ncol] = A[M,K](f16) @ Bpacked(f16) (+bias) ----------
// Each wave: one 16-row M tile x NT consecutive 16-col N tiles. K fully unrolled,
// A fragment reused across the NT B fragments (each a single 32B/lane vector load).
template <int K, int NT>
__global__ void gemm_wmma_kernel(const _Float16* __restrict__ A, const _Float16* __restrict__ Bp,
                                 const float* __restrict__ bias, float* __restrict__ C,
                                 int M, int Ncol) {
  int wave = (int)((blockIdx.x * blockDim.x + threadIdx.x) >> 5);
  int lane = threadIdx.x & 31;
  int ntiles = Ncol >> 4;
  int ngroups = ntiles / NT;
  int total = (M >> 4) * ngroups;
  if (wave >= total) return;  // wave-uniform: EXEC all-ones for WMMA
  int tm = wave / ngroups;
  int tng = wave % ngroups;
  int mrow = tm * 16 + (lane & 15);
  int khalf = (lane & 16) ? 8 : 0;
  v8f acc[NT];
#pragma unroll
  for (int nt = 0; nt < NT; ++nt) acc[nt] = (v8f){};
#pragma unroll
  for (int kb = 0; kb < K; kb += 32) {
    v16h a;
#pragma unroll
    for (int j = 0; j < 8; ++j) {
      int k0 = ((j < 4) ? (2 * j) : (2 * j + 8)) + khalf + kb;
      a[2 * j]     = A[(size_t)mrow * K + k0];
      a[2 * j + 1] = A[(size_t)mrow * K + k0 + 1];
    }
#pragma unroll
    for (int nt = 0; nt < NT; ++nt) {
      int frag = (kb >> 5) * ntiles + (tng * NT + nt);
      v16h b = *(const v16h*)(Bp + (size_t)frag * 512 + lane * 16);
      acc[nt] = __builtin_amdgcn_wmma_f32_16x16x32_f16(false, a, false, b, (short)0, acc[nt], false, false);
    }
  }
  int rowbase = tm * 16 + ((lane & 16) ? 8 : 0);
#pragma unroll
  for (int nt = 0; nt < NT; ++nt) {
    int ncol = (tng * NT + nt) * 16 + (lane & 15);
    float bs = bias ? bias[ncol] : 0.0f;
#pragma unroll
    for (int r = 0; r < 8; ++r) {
      C[(size_t)(rowbase + r) * Ncol + ncol] = acc[nt][r] + bs;
    }
  }
}

// ---------- edge pass 1: logits + segment max (uint-key atomicMax) ----------
__global__ void edge_logits_kernel(const int* __restrict__ ei, const float* __restrict__ ea,
                                   const float* __restrict__ q, const float* __restrict__ k,
                                   const float* __restrict__ Wedge, const float* __restrict__ att,
                                   float* __restrict__ logits, unsigned* __restrict__ mx) {
  int gid = blockIdx.x * blockDim.x + threadIdx.x;
  int e = gid >> 5;
  int lane = gid & 31;
  if (e >= EE) return;
  int src = ei[e];
  int dst = ei[EE + e];
  float ea0 = ea[e * 4 + 0], ea1 = ea[e * 4 + 1], ea2 = ea[e * 4 + 2], ea3 = ea[e * 4 + 3];
  float part[4] = {0.f, 0.f, 0.f, 0.f};
#pragma unroll
  for (int jj = 0; jj < 8; ++jj) {
    int idx = jj * 32 + lane;
    float ef = ea0 * Wedge[idx] + ea1 * Wedge[256 + idx] + ea2 * Wedge[512 + idx] + ea3 * Wedge[768 + idx];
    float z = q[(size_t)src * 256 + idx] + k[(size_t)dst * 256 + idx] + ef;
    z = (z > 0.f) ? z : 0.2f * z;            // leaky_relu(0.2)
    part[jj >> 1] += z * att[idx];           // att flat [4*64] == per-(head,dim)
  }
#pragma unroll
  for (int h = 0; h < 4; ++h) {
    float v = part[h];
    for (int off = 16; off > 0; off >>= 1) v += __shfl_xor(v, off);
    if (lane == 0) {
      logits[e * 4 + h] = v;
      atomicMax(&mx[dst * 4 + h], f2key(v));
    }
  }
}

// ---------- edge pass 2: w = exp(logit - max), segment sum ----------
__global__ void edge_weights_kernel(const int* __restrict__ ei, float* __restrict__ logits_w,
                                    const unsigned* __restrict__ mx, float* __restrict__ den) {
  int t = blockIdx.x * blockDim.x + threadIdx.x;
  if (t >= EE * 4) return;
  int e = t >> 2, h = t & 3;
  int dst = ei[EE + e];
  float m = key2f(mx[dst * 4 + h]);
  float wv = __expf(logits_w[t] - m);
  logits_w[t] = wv;
  atomicAdd(&den[dst * 4 + h], wv);
}

// ---------- edge pass 3: msg = a*(q[src]+e), scatter-add ----------
__global__ void edge_aggregate_kernel(const int* __restrict__ ei, const float* __restrict__ ea,
                                      const float* __restrict__ q, const float* __restrict__ w,
                                      const float* __restrict__ den, const float* __restrict__ Wedge,
                                      float* __restrict__ outacc) {
  int gid = blockIdx.x * blockDim.x + threadIdx.x;
  int e = gid >> 5;
  int lane = gid & 31;
  if (e >= EE) return;
  int src = ei[e];
  int dst = ei[EE + e];
  float ea0 = ea[e * 4 + 0], ea1 = ea[e * 4 + 1], ea2 = ea[e * 4 + 2], ea3 = ea[e * 4 + 3];
  float aH[4];
#pragma unroll
  for (int h = 0; h < 4; ++h) aH[h] = w[e * 4 + h] / (den[dst * 4 + h] + 1e-16f);
#pragma unroll
  for (int jj = 0; jj < 8; ++jj) {
    int idx = jj * 32 + lane;
    float ef = ea0 * Wedge[idx] + ea1 * Wedge[256 + idx] + ea2 * Wedge[512 + idx] + ea3 * Wedge[768 + idx];
    float msg = aH[jj >> 1] * (q[(size_t)src * 256 + idx] + ef);
    atomicAdd(&outacc[(size_t)dst * 256 + idx], msg);
  }
}

// ---------- masked select / masked alpha-accumulate into h_final ----------
__global__ void masked_copy_kernel(const float* __restrict__ srcbuf, float* __restrict__ dstbuf,
                                   const int* __restrict__ sensor, int mode) {
  int i = blockIdx.x * blockDim.x + threadIdx.x;
  if (i >= NN * 256) return;
  int s = sensor[i >> 8];
  bool m = (mode == 0) ? (s == 0) : ((mode == 1) ? (s == 1) : (s >= 2));
  if (m) dstbuf[i] = srcbuf[i];
}

__global__ void masked_axpy_kernel(const float* __restrict__ srcbuf, float* __restrict__ dstbuf,
                                   const int* __restrict__ sensor, int maskS,
                                   const int* __restrict__ gstep) {
  int i = blockIdx.x * blockDim.x + threadIdx.x;
  if (i >= NN * 256) return;
  if (sensor[i >> 8] != maskS) return;
  int step = gstep[0];
  float alpha = (step < 0) ? 0.0f : ((step >= 100) ? 0.5f : 0.5f * (float)step / 100.0f);
  dstbuf[i] += alpha * srcbuf[i];
}

// ---------- cross stats: per-dst count and sum(ea[:,0]) ----------
__global__ void cross_count_kernel(const int* __restrict__ ei, const float* __restrict__ ea,
                                   float* __restrict__ cnt, float* __restrict__ ssum) {
  int e = blockIdx.x * blockDim.x + threadIdx.x;
  if (e >= EE) return;
  int dst = ei[EE + e];
  atomicAdd(&cnt[dst], 1.0f);
  atomicAdd(&ssum[dst], ea[e * 4 + 0]);
}

// ---------- sigma-head extra input columns (rows 256.. of W1) ----------
__global__ void head_extra_kernel(float* __restrict__ hidden, const float* __restrict__ x,
                                  const float* __restrict__ W1x,
                                  const float* __restrict__ cnt, const float* __restrict__ ssum,
                                  int mode) {
  int i = blockIdx.x * blockDim.x + threadIdx.x;
  if (i >= NN * 128) return;
  int n = i >> 7, j = i & 127;
  float add;
  if (mode == 0) {  // lidar_sigma: x[:,0:2]
    add = x[n * 8 + 0] * W1x[j] + x[n * 8 + 1] * W1x[128 + j];
  } else {          // r*_sigma: x2, x3, mean_d, log1p(cnt)
    float c = cnt[n];
    float md = ssum[n] / fmaxf(c, 1.0f);
    float lc = log1pf(c);
    add = x[n * 8 + 2] * W1x[j] + x[n * 8 + 3] * W1x[128 + j] + md * W1x[256 + j] + lc * W1x[384 + j];
  }
  hidden[i] += add;
}

// ---------- head layer 2 + mask/bound epilogue (wave per node) ----------
__global__ void head_out_kernel(const float* __restrict__ hidden, const float* __restrict__ W2,
                                const float* __restrict__ b2, const int* __restrict__ sensor,
                                float* __restrict__ out, int outDim, int maskS,
                                int doBound, float lo, float hi) {
  int gid = blockIdx.x * blockDim.x + threadIdx.x;
  int n = gid >> 5;
  int lane = gid & 31;
  if (n >= NN) return;
  float hv[4];
#pragma unroll
  for (int r = 0; r < 4; ++r) hv[r] = fmaxf(hidden[(size_t)n * 128 + r * 32 + lane], 0.0f);
  bool m = (sensor[n] == maskS);
  for (int o = 0; o < outDim; ++o) {
    float v = 0.f;
#pragma unroll
    for (int r = 0; r < 4; ++r) v += hv[r] * W2[(r * 32 + lane) * outDim + o];
    for (int off = 16; off > 0; off >>= 1) v += __shfl_xor(v, off);
    if (lane == 0) {
      v += b2[o];
      float res;
      if (doBound) {
        float sg = 1.0f / (1.0f + __expf(-v));
        res = m ? (lo + (hi - lo) * sg) : lo;
      } else {
        res = m ? v : 0.0f;
      }
      out[(size_t)n * outDim + o] = res;
    }
  }
}

// =====================================================================
extern "C" void kernel_launch(void* const* d_in, const int* in_sizes, int n_in,
                              void* d_out, int out_size, void* d_ws, size_t ws_size,
                              hipStream_t stream) {
  const float* x      = (const float*)d_in[0];
  const int*   sensor = (const int*)d_in[1];
  const int*   gstep  = (const int*)d_in[18];

  // params flattened in insertion order starting at index 19
  // enc_lidar 19-22, enc_r1 23-26, enc_r2 27-30
  // st1_LL 31, st1_R1 35, st1_R2 39, st2_L2R1 43, st2_R12L 47, st2_L2R2 51, st2_R22L 55
  // adap_back.W 59, .b 60; heads 61..84
  const int relEi[7] = {4, 6, 8, 10, 12, 14, 16};
  const int relEa[7] = {5, 7, 9, 11, 13, 15, 17};
  const int relP[7]  = {31, 35, 39, 43, 47, 51, 55};
  const int st2mask[4] = {1, 0, 2, 0};  // L2R1->R1, R12L->L, L2R2->R2, R22L->L

  // ---- carve workspace ----
  char* p = (char*)d_ws;
  auto alloc = [&](size_t bytes) -> void* {
    void* r = (void*)p;
    p += (bytes + 255) & ~(size_t)255;
    return r;
  };
  float*    q       = (float*)alloc((size_t)NN * 256 * 4);
  float*    kbuf    = (float*)alloc((size_t)NN * 256 * 4);
  float*    tmp     = (float*)alloc((size_t)NN * 256 * 4);
  float*    hidden  = (float*)alloc((size_t)NN * 128 * 4);
  _Float16* h16     = (_Float16*)alloc((size_t)NN * 64 * 2);
  _Float16* hst116  = (_Float16*)alloc((size_t)NN * 256 * 2);
  _Float16* hbase16 = (_Float16*)alloc((size_t)NN * 64 * 2);
  _Float16* hfin16  = (_Float16*)alloc((size_t)NN * 256 * 2);
  float*    logitsw = (float*)alloc((size_t)EE * 4 * 4);
  unsigned* mx      = (unsigned*)alloc((size_t)NN * 4 * 4);
  float*    den     = (float*)alloc((size_t)NN * 4 * 4);
  float*    cnt1    = (float*)alloc((size_t)NN * 4);
  float*    sum1    = (float*)alloc((size_t)NN * 4);
  float*    cnt2    = (float*)alloc((size_t)NN * 4);
  float*    sum2    = (float*)alloc((size_t)NN * 4);
  _Float16* wsrc16[7];
  _Float16* wdst16[7];
  for (int r = 0; r < 7; ++r) {
    wsrc16[r] = (_Float16*)alloc(16384 * 2);
    wdst16[r] = (_Float16*)alloc(16384 * 2);
  }
  _Float16* adap16 = (_Float16*)alloc(16384 * 2);
  _Float16* w116[6];
  for (int i = 0; i < 6; ++i) w116[i] = (_Float16*)alloc(32768 * 2);

  // ---- launch helpers ----
  auto cvt = [&](const float* s, _Float16* d, int n) {
    f32_to_f16_kernel<<<(n + 255) / 256, 256, 0, stream>>>(s, d, n);
  };
  auto pack = [&](const float* B, _Float16* d, int K, int Ncol) {
    pack_b_kernel<<<(K * Ncol + 255) / 256, 256, 0, stream>>>(B, d, K, Ncol);
  };
  auto zf = [&](float* ptr, int n, float v) {
    zero_f32_kernel<<<(n + 255) / 256, 256, 0, stream>>>(ptr, n, v);
  };
  // K=64 GEMM (q/k projections): Ncol=256, 4 N-tiles per wave
  auto gemm64 = [&](const _Float16* A, const _Float16* Bp, const float* bias, float* C, int Ncol) {
    int waves = (NN / 16) * ((Ncol / 16) / 4);
    gemm_wmma_kernel<64, 4><<<(waves * 32 + 255) / 256, 256, 0, stream>>>(A, Bp, bias, C, NN, Ncol);
  };
  // K=256 GEMM (adap_back Ncol=64, heads Ncol=128): 4 N-tiles per wave
  auto gemm256 = [&](const _Float16* A, const _Float16* Bp, const float* bias, float* C, int Ncol) {
    int waves = (NN / 16) * ((Ncol / 16) / 4);
    gemm_wmma_kernel<256, 4><<<(waves * 32 + 255) / 256, 256, 0, stream>>>(A, Bp, bias, C, NN, Ncol);
  };

  // ---- prep: pack GEMM-side weights into f16 fragment-major layout ----
  for (int r = 0; r < 7; ++r) {
    pack((const float*)d_in[relP[r] + 0], wsrc16[r], 64, 256);
    pack((const float*)d_in[relP[r] + 1], wdst16[r], 64, 256);
  }
  pack((const float*)d_in[59], adap16, 256, 64);
  const int headP[6] = {61, 65, 69, 73, 77, 81};
  for (int i = 0; i < 6; ++i) pack((const float*)d_in[headP[i]], w116[i], 256, 128);

  // ---- encoders ----
  encoder_kernel<<<NN, 64, 0, stream>>>(
      x, sensor,
      (const float*)d_in[19], (const float*)d_in[20], (const float*)d_in[21], (const float*)d_in[22],
      (const float*)d_in[23], (const float*)d_in[24], (const float*)d_in[25], (const float*)d_in[26],
      (const float*)d_in[27], (const float*)d_in[28], (const float*)d_in[29], (const float*)d_in[30],
      h16);

  float* outH = (float*)d_out;  // h_final region [NN,256]

  // ---- one GAT relation ----
  auto gat = [&](int r, const _Float16* hin) {
    const int*   ei    = (const int*)d_in[relEi[r]];
    const float* ea    = (const float*)d_in[relEa[r]];
    const float* Wedge = (const float*)d_in[relP[r] + 2];
    const float* att   = (const float*)d_in[relP[r] + 3];
    gemm64(hin, wsrc16[r], nullptr, q, 256);
    gemm64(hin, wdst16[r], nullptr, kbuf, 256);
    zero_u32_kernel<<<(NN * 4 + 255) / 256, 256, 0, stream>>>(mx, NN * 4);
    zf(den, NN * 4, 0.0f);
    edge_logits_kernel<<<(EE * 32 + 255) / 256, 256, 0, stream>>>(ei, ea, q, kbuf, Wedge, att, logitsw, mx);
    edge_weights_kernel<<<(EE * 4 + 255) / 256, 256, 0, stream>>>(ei, logitsw, mx, den);
    zf(tmp, NN * 256, 0.0f);
    edge_aggregate_kernel<<<(EE * 32 + 255) / 256, 256, 0, stream>>>(ei, ea, q, logitsw, den, Wedge, tmp);
  };

  // ---- stage 1: LL / R1R1 / R2R2, masked-select into h_final ----
  for (int r = 0; r < 3; ++r) {
    gat(r, h16);
    masked_copy_kernel<<<(NN * 256 + 255) / 256, 256, 0, stream>>>(tmp, outH, sensor, r);
  }

  // ---- adap_back: h_base = h_st1 @ W + b ----
  cvt(outH, hst116, NN * 256);
  gemm256(hst116, adap16, (const float*)d_in[60], tmp, 64);
  cvt(tmp, hbase16, NN * 64);

  // ---- stage 2: cross relations, alpha-accumulated into h_final ----
  for (int r = 3; r < 7; ++r) {
    gat(r, hbase16);
    masked_axpy_kernel<<<(NN * 256 + 255) / 256, 256, 0, stream>>>(tmp, outH, sensor, st2mask[r - 3], gstep);
  }

  // ---- cross stats for L2R1 / L2R2 ----
  zf(cnt1, NN, 0.0f); zf(sum1, NN, 0.0f); zf(cnt2, NN, 0.0f); zf(sum2, NN, 0.0f);
  cross_count_kernel<<<(EE + 255) / 256, 256, 0, stream>>>((const int*)d_in[10], (const float*)d_in[11], cnt1, sum1);
  cross_count_kernel<<<(EE + 255) / 256, 256, 0, stream>>>((const int*)d_in[14], (const float*)d_in[15], cnt2, sum2);

  // ---- heads ----
  cvt(outH, hfin16, NN * 256);
  const float MIN_LOG_LIDAR = -5.2983174f;  // log(0.005)
  const float MIN_LOG_RADAR = -3.9120230f;  // log(0.02)
  float* o_lmu  = outH + (size_t)NN * 256;
  float* o_lls  = o_lmu + (size_t)NN * 2;
  float* o_r1mu = o_lls + (size_t)NN * 2;
  float* o_r1ls = o_r1mu + (size_t)NN;
  float* o_r2mu = o_r1ls + (size_t)NN;
  float* o_r2ls = o_r2mu + (size_t)NN;

  struct HeadCfg {
    int pidx, outDim, maskS, doBound, extraMode;  // extraMode: -1 none, 0 lidar, 1 radar
    float lo;
    const float *cnt, *sum;
    float* out;
  };
  HeadCfg heads[6] = {
      {61, 2, 0, 0, -1, 0.0f,          nullptr, nullptr, o_lmu},
      {65, 1, 1, 0, -1, 0.0f,          nullptr, nullptr, o_r1mu},
      {69, 1, 2, 0, -1, 0.0f,          nullptr, nullptr, o_r2mu},
      {73, 2, 0, 1, 0,  MIN_LOG_LIDAR, nullptr, nullptr, o_lls},
      {77, 1, 1, 1, 1,  MIN_LOG_RADAR, cnt1,    sum1,    o_r1ls},
      {81, 1, 2, 1, 1,  MIN_LOG_RADAR, cnt2,    sum2,    o_r2ls},
  };
  for (int i = 0; i < 6; ++i) {
    HeadCfg& H = heads[i];
    const float* b1 = (const float*)d_in[H.pidx + 1];
    const float* W2 = (const float*)d_in[H.pidx + 2];
    const float* b2 = (const float*)d_in[H.pidx + 3];
    gemm256(hfin16, w116[i], b1, hidden, 128);
    if (H.extraMode >= 0) {
      const float* W1x = (const float*)d_in[H.pidx] + 256 * 128;  // rows 256.. of W1
      head_extra_kernel<<<(NN * 128 + 255) / 256, 256, 0, stream>>>(hidden, x, W1x, H.cnt, H.sum, H.extraMode);
    }
    head_out_kernel<<<(NN * 32 + 255) / 256, 256, 0, stream>>>(
        hidden, W2, b2, sensor, H.out, H.outDim, H.maskS, H.doBound, H.lo, 2.0f);
  }
}
